// GAT_23888608101379
// MI455X (gfx1250) — matrix operand order repaired and checked by reading, not compiled, not run on hardware
//
#include <hip/hip_runtime.h>
#include <math.h>

#define NNODES 50000
#define NEDGES 800000
#define ETOT   (NEDGES + NNODES)
#define CDIM   64

typedef __attribute__((ext_vector_type(16))) __bf16 v16bf;
typedef __attribute__((ext_vector_type(8)))  float  v8f;

// ---------- monotonic float <-> uint key (for atomicMax on floats incl. negatives) ----------
__device__ __forceinline__ unsigned int f2key(float f) {
  unsigned int b = __float_as_uint(f);
  return (b & 0x80000000u) ? ~b : (b | 0x80000000u);
}
__device__ __forceinline__ float key2f(unsigned int k) {
  unsigned int b = (k & 0x80000000u) ? (k ^ 0x80000000u) : ~k;
  return __uint_as_float(b);
}

__device__ __forceinline__ void edge_endpoints(const long long* __restrict__ ei, int e,
                                               int& s, int& d) {
  if (e < NEDGES) {
    s = (int)ei[e];
    d = (int)ei[NEDGES + e];
  } else {            // implicit self-loops appended after the real edges
    s = d = e - NEDGES;
  }
}

// ---------------- f32 -> bf16 elementwise conversion (input features, layer 0 only) ---------
__global__ void convert_f32_bf16(const float* __restrict__ src, __bf16* __restrict__ dst,
                                 long n) {
  long t = (long)blockIdx.x * blockDim.x + threadIdx.x;
  if (t >= n) return;
  dst[t] = (__bf16)src[t];
}

// ---------------- W[K,Fout] f32 -> Wt[Fout,K] bf16 (transpose + convert) -------------------
__global__ void convert_w_t(const float* __restrict__ W, __bf16* __restrict__ Wt,
                            int K, int Fout) {
  int t = blockIdx.x * blockDim.x + threadIdx.x;
  if (t >= K * Fout) return;
  int col = t / K, k = t - col * K;
  Wt[t] = (__bf16)W[(long)k * Fout + col];
}

// ---------------- GEMM: H[N,Fout] = A[N,K] @ Wt[Fout,K]^T, bf16 WMMA, f32 accum ------------
// One wave32 computes one 16x16 output tile. Both operands are packed bf16: per lane per
// K-step, two b128 loads each. K is a template constant -> fully unrolled WMMA chain.
template <int K>
__global__ void __launch_bounds__(128)
gemm_bf16_wmma(const __bf16* __restrict__ A, const __bf16* __restrict__ Wt,
               float* __restrict__ H, int n, int Fout) {
  const int lane  = threadIdx.x & 31;
  const int wave  = blockIdx.x * (blockDim.x >> 5) + (threadIdx.x >> 5);
  const int tilesN = Fout >> 4;
  const int tM = wave / tilesN;           // wave-uniform -> EXEC stays all-ones for WMMA
  const int tN = wave - tM * tilesN;
  if (tM * 16 >= n) return;
  const int hlf = lane >> 4;              // 0: lanes 0-15, 1: lanes 16-31
  const int l15 = lane & 15;

  // A 16x32 bf16 layout: lanes 0-15 hold K {0..7,16..23}, lanes 16-31 hold K {8..15,24..31}
  const __bf16* arow = A  + (long)(tM * 16 + l15) * K + hlf * 8;
  // B 32x16 bf16 layout: lane = column; lanes 0-15 hold K {0..7,16..23}, lanes 16-31 K {8..15,24..31}
  const __bf16* bcol = Wt + (long)(tN * 16 + l15) * K + hlf * 8;

  v8f acc = {};
#pragma unroll
  for (int k0 = 0; k0 < K; k0 += 32) {
    v16bf a, b;
    ((uint4*)&a)[0] = *(const uint4*)(arow + k0);        // K off .. off+7
    ((uint4*)&a)[1] = *(const uint4*)(arow + k0 + 16);   // K off+16 .. off+23
    ((uint4*)&b)[0] = *(const uint4*)(bcol + k0);        // K off .. off+7
    ((uint4*)&b)[1] = *(const uint4*)(bcol + k0 + 16);   // K off+16 .. off+23
    acc = __builtin_amdgcn_wmma_f32_16x16x32_bf16(false, a, false, b,
                                                  (short)0, acc, false, false);
  }
  // D layout: lane -> col (lane&15); VGPR r -> row r + 8*(lane>=16)
  float* hp = H + (long)(tM * 16 + hlf * 8) * Fout + tN * 16 + l15;
#pragma unroll
  for (int r = 0; r < 8; ++r) hp[(long)r * Fout] = acc[r];
}

// ---------------- alpha_s / alpha_d : per (node, head) dot with attention vectors ----------
__global__ void alpha_kernel(const float* __restrict__ Hf, const float* __restrict__ as,
                             const float* __restrict__ ad, float* __restrict__ alphas,
                             float* __restrict__ alphad, int heads) {
  int t = blockIdx.x * blockDim.x + threadIdx.x;
  if (t >= NNODES * heads) return;
  int nidx = t / heads, h = t - nidx * heads;
  const float* hp = Hf + (long)nidx * heads * CDIM + h * CDIM;
  const float* sp = as + h * CDIM;
  const float* dp = ad + h * CDIM;
  float ss = 0.f, sd = 0.f;
#pragma unroll 8
  for (int c = 0; c < CDIM; ++c) { float v = hp[c]; ss += v * sp[c]; sd += v * dp[c]; }
  alphas[t] = ss;
  alphad[t] = sd;
}

// ---------------- edge pass 1: segment max of leaky_relu(as[src]+ad[dst]) ------------------
__global__ void edge_max_kernel(const long long* __restrict__ ei,
                                const float* __restrict__ as, const float* __restrict__ ad,
                                unsigned int* __restrict__ mkey, int heads) {
  long t = (long)blockIdx.x * blockDim.x + threadIdx.x;
  if (t >= (long)ETOT * heads) return;
  int e = (int)(t / heads), h = (int)(t - (long)e * heads);
  int s, d; edge_endpoints(ei, e, s, d);
  float v = as[(long)s * heads + h] + ad[(long)d * heads + h];
  v = v > 0.f ? v : 0.2f * v;
  atomicMax(&mkey[(long)d * heads + h], f2key(v));
}

// ---------------- edge pass 2: exp(e - max) and segment-sum of denominators ----------------
__global__ void edge_exp_kernel(const long long* __restrict__ ei,
                                const float* __restrict__ as, const float* __restrict__ ad,
                                const unsigned int* __restrict__ mkey,
                                float* __restrict__ exb, float* __restrict__ denom, int heads) {
  long t = (long)blockIdx.x * blockDim.x + threadIdx.x;
  if (t >= (long)ETOT * heads) return;
  int e = (int)(t / heads), h = (int)(t - (long)e * heads);
  int s, d; edge_endpoints(ei, e, s, d);
  float v = as[(long)s * heads + h] + ad[(long)d * heads + h];
  v = v > 0.f ? v : 0.2f * v;
  float m  = key2f(mkey[(long)d * heads + h]);
  float ex = __expf(v - m);
  exb[t] = ex;
  atomicAdd(&denom[(long)d * heads + h], ex);
}

// ---------------- edge pass 3: acc[dst] += (ex/denom) * h[src], 4 channels/thread ----------
__global__ void edge_aggr_kernel(const long long* __restrict__ ei,
                                 const float* __restrict__ exb, const float* __restrict__ denom,
                                 const float* __restrict__ Hf, float* __restrict__ acc,
                                 int heads) {
  const int groups = (heads * CDIM) >> 2;      // 4 channels per thread
  long t = (long)blockIdx.x * blockDim.x + threadIdx.x;
  if (t >= (long)ETOT * groups) return;
  int e  = (int)(t / groups);
  int g  = (int)(t - (long)e * groups);
  int c0 = g << 2;
  int h  = c0 >> 6;                            // CDIM == 64
  int s, d; edge_endpoints(ei, e, s, d);
  float w = exb[(long)e * heads + h] / (denom[(long)d * heads + h] + 1e-16f);
  const float4 hv = *(const float4*)(Hf + (long)s * heads * CDIM + c0);
  float* ap = acc + (long)d * heads * CDIM + c0;
  atomicAdd(ap + 0, w * hv.x);
  atomicAdd(ap + 1, w * hv.y);
  atomicAdd(ap + 2, w * hv.z);
  atomicAdd(ap + 3, w * hv.w);
}

// ------- finalize (layers 0/1): next-layer act (bf16) = relu(acc + bias), fused convert ----
__global__ void finalize_relu_bf16(const float* __restrict__ acc, const float* __restrict__ bias,
                                   __bf16* __restrict__ out, int F) {
  long t = (long)blockIdx.x * blockDim.x + threadIdx.x;
  if (t >= (long)NNODES * F) return;
  int j = (int)(t % F);
  float v = acc[t] + bias[j];
  out[t] = (__bf16)(v > 0.f ? v : 0.f);
}

// ---------------- layer-2 finalize fused with global mean reduction -----------------------
__global__ void finalize_mean(const float* __restrict__ acc, const float* __restrict__ bias,
                              float* __restrict__ meanacc) {
  __shared__ float sm[256];
  long t = (long)blockIdx.x * blockDim.x + threadIdx.x;   // 4 nodes x 64 channels / block
  float v = 0.f;
  if (t < (long)NNODES * CDIM) {
    int j = (int)(t & 63);
    v = acc[t] + bias[j];
  }
  sm[threadIdx.x] = v;
  __syncthreads();
  if (threadIdx.x < 64) {
    float s = sm[threadIdx.x] + sm[threadIdx.x + 64] + sm[threadIdx.x + 128] + sm[threadIdx.x + 192];
    atomicAdd(&meanacc[threadIdx.x], s);
  }
}

// ---------------- prediction head: out = (meanacc/N) @ hw + hb ------------------------------
__global__ void head_kernel(const float* __restrict__ meanacc, const float* __restrict__ hw,
                            const float* __restrict__ hb, float* __restrict__ out) {
  int j = threadIdx.x;                     // 64 threads
  float s = hb[j];
  const float inv = 1.0f / (float)NNODES;
#pragma unroll 8
  for (int i = 0; i < CDIM; ++i) s += (meanacc[i] * inv) * hw[i * CDIM + j];
  out[j] = s;
}

// -------------------------------------------------------------------------------------------
static void run_gat_layer(const __bf16* actBf, int Fin,
                          const float* Wl, const float* as, const float* ad,
                          int heads, __bf16* wtBf, float* hbuf, float* accbuf,
                          float* alphaS, float* alphaD,
                          unsigned int* mkey, float* denom, float* exb,
                          const long long* ei, hipStream_t stream) {
  const int Fout = heads * CDIM;
  {
    int total = Fin * Fout;
    convert_w_t<<<(total + 255) / 256, 256, 0, stream>>>(Wl, wtBf, Fin, Fout);
  }
  {
    long waves  = (long)(NNODES / 16) * (Fout / 16);
    int  blocks = (int)((waves + 3) / 4);
    if (Fin == 128)
      gemm_bf16_wmma<128><<<blocks, 128, 0, stream>>>(actBf, wtBf, hbuf, NNODES, Fout);
    else
      gemm_bf16_wmma<256><<<blocks, 128, 0, stream>>>(actBf, wtBf, hbuf, NNODES, Fout);
  }
  {
    int total = NNODES * heads;
    alpha_kernel<<<(total + 255) / 256, 256, 0, stream>>>(hbuf, as, ad, alphaS, alphaD, heads);
  }
  hipMemsetAsync(mkey,   0, (size_t)NNODES * heads * sizeof(unsigned int), stream);
  hipMemsetAsync(denom,  0, (size_t)NNODES * heads * sizeof(float), stream);
  hipMemsetAsync(accbuf, 0, (size_t)NNODES * Fout * sizeof(float), stream);
  {
    long total = (long)ETOT * heads;
    int  blocks = (int)((total + 255) / 256);
    edge_max_kernel<<<blocks, 256, 0, stream>>>(ei, alphaS, alphaD, mkey, heads);
    edge_exp_kernel<<<blocks, 256, 0, stream>>>(ei, alphaS, alphaD, mkey, exb, denom, heads);
  }
  {
    long total = (long)ETOT * (Fout >> 2);
    int  blocks = (int)((total + 255) / 256);
    edge_aggr_kernel<<<blocks, 256, 0, stream>>>(ei, exb, denom, hbuf, accbuf, heads);
  }
}

extern "C" void kernel_launch(void* const* d_in, const int* in_sizes, int n_in,
                              void* d_out, int out_size, void* d_ws, size_t ws_size,
                              hipStream_t stream) {
  (void)in_sizes; (void)n_in; (void)out_size; (void)ws_size;
  // ---- inputs (setup_inputs order) ----
  const float*     x   = (const float*)d_in[0];
  const long long* ei  = (const long long*)d_in[1];   // int64 [2,E]
  const float* W0  = (const float*)d_in[2];
  const float* a0s = (const float*)d_in[3];
  const float* a0d = (const float*)d_in[4];
  const float* b0  = (const float*)d_in[5];
  const float* W1  = (const float*)d_in[6];
  const float* a1s = (const float*)d_in[7];
  const float* a1d = (const float*)d_in[8];
  const float* b1  = (const float*)d_in[9];
  const float* W2  = (const float*)d_in[10];
  const float* a2s = (const float*)d_in[11];
  const float* a2d = (const float*)d_in[12];
  const float* b2  = (const float*)d_in[13];
  const float* hw  = (const float*)d_in[14];
  const float* hb  = (const float*)d_in[15];
  float* out = (float*)d_out;

  // ---- workspace carve (~145 MB total; everything hot fits the 192 MB L2) ----
  char* p = (char*)d_ws;
  float*  hBuf   = (float*)p;  p += (size_t)NNODES * 256 * sizeof(float);   // 51.2 MB
  float*  accBuf = (float*)p;  p += (size_t)NNODES * 256 * sizeof(float);   // 51.2 MB
  __bf16* actBf  = (__bf16*)p; p += (size_t)NNODES * 256 * sizeof(__bf16);  // 25.6 MB
  __bf16* wtBf   = (__bf16*)p; p += (size_t)256 * 256 * sizeof(__bf16);     // 128 KB
  float* alphaS  = (float*)p;  p += (size_t)NNODES * 4 * sizeof(float);
  float* alphaD  = (float*)p;  p += (size_t)NNODES * 4 * sizeof(float);
  unsigned int* mkey = (unsigned int*)p; p += (size_t)NNODES * 4 * sizeof(unsigned int);
  float* denom   = (float*)p;  p += (size_t)NNODES * 4 * sizeof(float);
  float* exb     = (float*)p;  p += (size_t)ETOT * 4 * sizeof(float);       // 13.6 MB
  float* meanAcc = (float*)p;  p += 64 * sizeof(float);

  // ---- layer 0: 128 -> 4x64, relu ----
  convert_f32_bf16<<<((long)NNODES * 128 + 255) / 256, 256, 0, stream>>>(x, actBf, (long)NNODES * 128);
  run_gat_layer(actBf, 128, W0, a0s, a0d, 4, wtBf, hBuf, accBuf,
                alphaS, alphaD, mkey, denom, exb, ei, stream);
  finalize_relu_bf16<<<(NNODES * 256 + 255) / 256, 256, 0, stream>>>(accBuf, b0, actBf, 256);

  // ---- layer 1: 256 -> 4x64, relu ----
  run_gat_layer(actBf, 256, W1, a1s, a1d, 4, wtBf, hBuf, accBuf,
                alphaS, alphaD, mkey, denom, exb, ei, stream);
  finalize_relu_bf16<<<(NNODES * 256 + 255) / 256, 256, 0, stream>>>(accBuf, b1, actBf, 256);

  // ---- layer 2: 256 -> 1x64 ----
  run_gat_layer(actBf, 256, W2, a2s, a2d, 1, wtBf, hBuf, accBuf,
                alphaS, alphaD, mkey, denom, exb, ei, stream);

  // ---- global mean + head ----
  hipMemsetAsync(meanAcc, 0, 64 * sizeof(float), stream);
  finalize_mean<<<(NNODES * CDIM + 255) / 256, 256, 0, stream>>>(accBuf, b2, meanAcc);
  head_kernel<<<1, 64, 0, stream>>>(meanAcc, hw, hb, out);
}
